// PAWSA_395136991260
// MI455X (gfx1250) — matrix operand order repaired and checked by reading, not compiled, not run on hardware
//
#include <hip/hip_runtime.h>

typedef __bf16 bf16_t;
typedef __bf16 v2bf  __attribute__((ext_vector_type(2)));
typedef __bf16 v8bf  __attribute__((ext_vector_type(8)));
typedef __bf16 v16bf __attribute__((ext_vector_type(16)));
typedef float  v8f   __attribute__((ext_vector_type(8)));

#define NTOK   49
#define DIM    512
#define HEADS  16
#define HDIM   32
#define BWIN   2048
#define MTOT   (BWIN * NTOK)   // 100352 = 784 * 128

// ---- CDNA5 async global->LDS path (guarded so compile never breaks) ----
#if defined(__AMDGCN__) && __has_builtin(__builtin_amdgcn_global_load_async_to_lds_b128)
#define HAS_ASYNC_LDS 1
typedef int v4i_a __attribute__((ext_vector_type(4)));
typedef __attribute__((address_space(1))) v4i_a* gas_ptr;   // global int4*
typedef __attribute__((address_space(3))) v4i_a* las_ptr;   // LDS int4*
__device__ __forceinline__ void async_wait0() {
#if __has_builtin(__builtin_amdgcn_s_wait_asynccnt)
    __builtin_amdgcn_s_wait_asynccnt(0);
#else
    asm volatile("s_wait_asynccnt 0" ::: "memory");
#endif
}
#else
#define HAS_ASYNC_LDS 0
#endif

// Concatenate two 16-byte runs of bf16 into one 16-element WMMA fragment.
__device__ __forceinline__ v16bf frag_cat(const bf16_t* p0, const bf16_t* p1) {
    union { struct { v8bf lo, hi; } s; v16bf v; } u;
    u.s.lo = *(const v8bf*)p0;
    u.s.hi = *(const v8bf*)p1;
    return u.v;
}

// ---------------------------------------------------------------------------
// Tiled GEMM:  Out[M x Nc] = A[M x K] * Bw[K x Nc] + bias
//   BM=128, BN=128, BK=32. 256 threads = 8 waves; wave owns 32x64 sub-tile.
//   Double-buffered LDS; next tile's global loads are issued before the
//   current tile's WMMAs. bf16 A tiles go through async global->LDS DMA.
// ---------------------------------------------------------------------------
template<bool A_BF16, bool OUT_BF16>
__global__ __launch_bounds__(256)
void gemm_bf16_wmma(const void* __restrict__ Ap,
                    const float* __restrict__ Bw,
                    const float* __restrict__ bias,
                    void* __restrict__ Outp,
                    int Nc, int K)
{
    __shared__ __align__(16) bf16_t sA[2][128][40];
    __shared__ __align__(16) bf16_t sB[2][128][40];

    const int m0   = blockIdx.y << 7;
    const int n0   = blockIdx.x << 7;
    const int tid  = threadIdx.x;
    const int lane = tid & 31;
    const int wave = tid >> 5;
    const int wr   = wave >> 1;      // 0..3 : 32-row group
    const int wc   = wave & 1;       // 0..1 : 64-col group
    const int half = lane >> 4;
    const int l16  = lane & 15;

    // staging coordinates
    const int arow = tid >> 1;           // 0..127
    const int akh  = (tid & 1) << 4;     // 0 / 16
    const int bkk  = (tid >> 4) << 1;    // 0,2,..,30  (k pair)
    const int bnb  = (tid & 15) << 3;    // 0..120     (8 cols)

    v8f acc[2][4];
    #pragma unroll
    for (int mi = 0; mi < 2; ++mi)
        #pragma unroll
        for (int ni = 0; ni < 4; ++ni)
            acc[mi][ni] = {};

    const int kTiles = K >> 5;

    float ta[16];      // f32 A staging regs
    v8bf  tab[2];      // bf16 A staging regs (non-async fallback)
    float tb[16];      // B staging regs (rows bkk, bkk+1)

    // ---------------- prologue: stage tile 0 into buffer 0 ----------------
    {
        if (A_BF16) {
            const bf16_t* src = (const bf16_t*)Ap + (size_t)(m0 + arow) * K + akh;
#if HAS_ASYNC_LDS
            __builtin_amdgcn_global_load_async_to_lds_b128(
                (gas_ptr)src, (las_ptr)&sA[0][arow][akh], 0, 0);
            __builtin_amdgcn_global_load_async_to_lds_b128(
                (gas_ptr)(src + 8), (las_ptr)&sA[0][arow][akh + 8], 0, 0);
#else
            *(v8bf*)&sA[0][arow][akh]     = *(const v8bf*)src;
            *(v8bf*)&sA[0][arow][akh + 8] = *(const v8bf*)(src + 8);
#endif
        } else {
            const float* src = (const float*)Ap + (size_t)(m0 + arow) * K + akh;
            __align__(16) float t[16];
            *(float4*)&t[0]  = *(const float4*)(src + 0);
            *(float4*)&t[4]  = *(const float4*)(src + 4);
            *(float4*)&t[8]  = *(const float4*)(src + 8);
            *(float4*)&t[12] = *(const float4*)(src + 12);
            union { bf16_t b[16]; v8bf v[2]; } pk;
            #pragma unroll
            for (int i = 0; i < 16; ++i) pk.b[i] = (bf16_t)t[i];
            *(v8bf*)&sA[0][arow][akh]     = pk.v[0];
            *(v8bf*)&sA[0][arow][akh + 8] = pk.v[1];
        }
        const float* s0 = Bw + (size_t)bkk * Nc + n0 + bnb;
        const float* s1 = s0 + Nc;
        __align__(16) float t[16];
        *(float4*)&t[0]  = *(const float4*)(s0 + 0);
        *(float4*)&t[4]  = *(const float4*)(s0 + 4);
        *(float4*)&t[8]  = *(const float4*)(s1 + 0);
        *(float4*)&t[12] = *(const float4*)(s1 + 4);
        #pragma unroll
        for (int i = 0; i < 8; ++i) {
            v2bf p; p[0] = (bf16_t)t[i]; p[1] = (bf16_t)t[8 + i];
            *(v2bf*)&sB[0][bnb + i][bkk] = p;
        }
    }
#if HAS_ASYNC_LDS
    if (A_BF16) async_wait0();
#endif
    __syncthreads();

    // ---------------- main loop ----------------
    for (int kt = 0; kt < kTiles; ++kt) {
        const int cur = kt & 1;
        const int nxt = cur ^ 1;
        const bool hasNext = (kt + 1) < kTiles;
        const int k1 = (kt + 1) << 5;

        // issue next tile's global loads before computing the current tile
        if (hasNext) {
            if (A_BF16) {
                const bf16_t* src = (const bf16_t*)Ap + (size_t)(m0 + arow) * K + k1 + akh;
#if HAS_ASYNC_LDS
                __builtin_amdgcn_global_load_async_to_lds_b128(
                    (gas_ptr)src, (las_ptr)&sA[nxt][arow][akh], 0, 0);
                __builtin_amdgcn_global_load_async_to_lds_b128(
                    (gas_ptr)(src + 8), (las_ptr)&sA[nxt][arow][akh + 8], 0, 0);
#else
                tab[0] = *(const v8bf*)src;
                tab[1] = *(const v8bf*)(src + 8);
#endif
            } else {
                const float* src = (const float*)Ap + (size_t)(m0 + arow) * K + k1 + akh;
                *(float4*)&ta[0]  = *(const float4*)(src + 0);
                *(float4*)&ta[4]  = *(const float4*)(src + 4);
                *(float4*)&ta[8]  = *(const float4*)(src + 8);
                *(float4*)&ta[12] = *(const float4*)(src + 12);
            }
            const float* s0 = Bw + (size_t)(k1 + bkk) * Nc + n0 + bnb;
            const float* s1 = s0 + Nc;
            *(float4*)&tb[0]  = *(const float4*)(s0 + 0);
            *(float4*)&tb[4]  = *(const float4*)(s0 + 4);
            *(float4*)&tb[8]  = *(const float4*)(s1 + 0);
            *(float4*)&tb[12] = *(const float4*)(s1 + 4);
        }

        // ---- compute: 8 WMMAs per wave on the current LDS buffer ----
        v16bf af[2];
        #pragma unroll
        for (int mi = 0; mi < 2; ++mi) {
            const int row = (wr << 5) + (mi << 4) + l16;
            af[mi] = frag_cat(&sA[cur][row][half * 8], &sA[cur][row][16 + half * 8]);
        }
        #pragma unroll
        for (int ni = 0; ni < 4; ++ni) {
            const int col = (wc << 6) + (ni << 4) + l16;
            v16bf bfrag = frag_cat(&sB[cur][col][half * 8], &sB[cur][col][16 + half * 8]);
            #pragma unroll
            for (int mi = 0; mi < 2; ++mi)
                acc[mi][ni] = __builtin_amdgcn_wmma_f32_16x16x32_bf16(
                    false, af[mi], false, bfrag, (short)0, acc[mi][ni], false, false);
        }

        // drain staged registers into the other LDS buffer
        if (hasNext) {
            if (A_BF16) {
#if !HAS_ASYNC_LDS
                *(v8bf*)&sA[nxt][arow][akh]     = tab[0];
                *(v8bf*)&sA[nxt][arow][akh + 8] = tab[1];
#endif
            } else {
                union { bf16_t b[16]; v8bf v[2]; } pk;
                #pragma unroll
                for (int i = 0; i < 16; ++i) pk.b[i] = (bf16_t)ta[i];
                *(v8bf*)&sA[nxt][arow][akh]     = pk.v[0];
                *(v8bf*)&sA[nxt][arow][akh + 8] = pk.v[1];
            }
            #pragma unroll
            for (int i = 0; i < 8; ++i) {
                v2bf p; p[0] = (bf16_t)tb[i]; p[1] = (bf16_t)tb[8 + i];
                *(v2bf*)&sB[nxt][bnb + i][bkk] = p;
            }
        }
#if HAS_ASYNC_LDS
        if (A_BF16 && hasNext) async_wait0();
#endif
        __syncthreads();
    }

    // ---------------- epilogue: bias add + store ----------------
    #pragma unroll
    for (int mi = 0; mi < 2; ++mi) {
        #pragma unroll
        for (int ni = 0; ni < 4; ++ni) {
            const int col = n0 + (wc << 6) + (ni << 4) + l16;
            const float bv = bias[col];
            #pragma unroll
            for (int j = 0; j < 8; ++j) {
                const int row = m0 + (wr << 5) + (mi << 4) + half * 8 + j;
                const float v = acc[mi][ni][j] + bv;
                if (OUT_BF16)
                    ((bf16_t*)Outp)[(size_t)row * Nc + col] = (bf16_t)v;
                else
                    ((float*)Outp)[(size_t)row * Nc + col] = v;
            }
        }
    }
}

// ---------------------------------------------------------------------------
// Pre-pass: materialize padded relative-position bias  bias_pad[16][64][64]
// (-1e30 in masked region -> branch-free softmax in the attention kernel).
// ---------------------------------------------------------------------------
__global__ __launch_bounds__(256)
void bias_pad_kernel(const float* __restrict__ rpb,
                     const int*   __restrict__ relidx,
                     float* __restrict__ bias_pad)
{
    const int head = blockIdx.x;
    for (int i = threadIdx.x; i < 64 * 64; i += 256) {
        const int rr = i >> 6, cm = i & 63;
        float v = -1e30f;
        if (rr < NTOK && cm < NTOK)
            v = rpb[relidx[rr * NTOK + cm] * HEADS + head];
        bias_pad[head * 4096 + i] = v;
    }
}

// ---------------------------------------------------------------------------
// Window attention: one block per (window, head). 128 threads = 4 waves,
// each wave owns a 16-row strip of the (padded) 64x64 score matrix.
//   S = (Q * scale) K^T + bias ; softmax ; O = P V ; stored bf16 to ws.
// ---------------------------------------------------------------------------
__global__ __launch_bounds__(128)
void attn_wmma(const bf16_t* __restrict__ qkv,
               const float*  __restrict__ bias_pad,
               bf16_t* __restrict__ attn_out)
{
    __shared__ __align__(16) bf16_t kS[64][40];     // K rows [token][d]
    __shared__ __align__(16) bf16_t vS[32][72];     // V transposed [d][token]
    __shared__ __align__(16) bf16_t pS[4][16][72];  // per-wave P strip 16x64

    const int b    = blockIdx.x >> 4;
    const int head = blockIdx.x & 15;
    const int tid  = threadIdx.x;
    const int lane = tid & 31;
    const int wave = tid >> 5;
    const int half = lane >> 4;
    const int l16  = lane & 15;
    const float scale = 0.17677669529663688f;   // 1/sqrt(32)

    // ---- stage K (row-major) and V (transposed) for this (window, head) ----
    {
        const int m  = tid >> 1;                 // token 0..63
        const int dh = (tid & 1) << 4;           // d half 0 / 16
        union { bf16_t bb[16]; v8bf v[2]; } kk{}, vv{};
        if (m < NTOK) {
            const bf16_t* kp = qkv + (size_t)(b * NTOK + m) * (3 * DIM)
                                   + DIM + head * HDIM + dh;
            const bf16_t* vp = kp + DIM;
            kk.v[0] = *(const v8bf*)kp;  kk.v[1] = *(const v8bf*)(kp + 8);
            vv.v[0] = *(const v8bf*)vp;  vv.v[1] = *(const v8bf*)(vp + 8);
        }
        *(v8bf*)&kS[m][dh]     = kk.v[0];
        *(v8bf*)&kS[m][dh + 8] = kk.v[1];
        #pragma unroll
        for (int i = 0; i < 16; ++i) vS[dh + i][m] = vv.bb[i];
    }
    __syncthreads();

    const int rbase = wave << 4;

    // ---- Q fragment straight from global bf16 workspace ----
    v16bf qf = {};
    {
        const int n = rbase + l16;
        if (n < NTOK) {
            const bf16_t* qp = qkv + (size_t)(b * NTOK + n) * (3 * DIM)
                                   + head * HDIM + half * 8;
            qf = frag_cat(qp, qp + 16);
        }
    }

    // ---- S = Q K^T  (4 WMMAs: 16x64 strip, K=32) ----
    v8f sacc[4];
    #pragma unroll
    for (int nt = 0; nt < 4; ++nt) sacc[nt] = {};
    #pragma unroll
    for (int nt = 0; nt < 4; ++nt) {
        const int m = (nt << 4) + l16;
        v16bf kf = frag_cat(&kS[m][half * 8], &kS[m][16 + half * 8]);
        sacc[nt] = __builtin_amdgcn_wmma_f32_16x16x32_bf16(
            false, qf, false, kf, (short)0, sacc[nt], false, false);
    }

    // ---- branch-free bias + softmax (row stats via shfl in 16-lane groups) ----
    const float* bias_head = bias_pad + head * 4096;
    float rinv[8];
    #pragma unroll
    for (int j = 0; j < 8; ++j) {
        const int rr = rbase + half * 8 + j;           // query token row
        const float* brow = bias_head + rr * 64 + l16; // coalesced, L2-resident
        float pv[4];
        float mx = -1e30f;
        #pragma unroll
        for (int nt = 0; nt < 4; ++nt) {
            const float s = sacc[nt][j] * scale + brow[nt * 16];
            pv[nt] = s;
            mx = fmaxf(mx, s);
        }
        mx = fmaxf(mx, __shfl_xor(mx, 1, 32));
        mx = fmaxf(mx, __shfl_xor(mx, 2, 32));
        mx = fmaxf(mx, __shfl_xor(mx, 4, 32));
        mx = fmaxf(mx, __shfl_xor(mx, 8, 32));
        float sum = 0.f;
        #pragma unroll
        for (int nt = 0; nt < 4; ++nt) {
            const float p = __expf(pv[nt] - mx);       // masked entries -> ~0
            sum += p;
            pS[wave][half * 8 + j][(nt << 4) + l16] = (bf16_t)p;
        }
        sum += __shfl_xor(sum, 1, 32);
        sum += __shfl_xor(sum, 2, 32);
        sum += __shfl_xor(sum, 4, 32);
        sum += __shfl_xor(sum, 8, 32);
        rinv[j] = (sum > 0.f) ? (1.f / sum) : 0.f;
    }

    // ---- O = P V  (2 K-halves x 2 d-tiles = 4 WMMAs) ----
    v8f oacc[2];
    oacc[0] = {};  oacc[1] = {};
    #pragma unroll
    for (int kk2 = 0; kk2 < 2; ++kk2) {
        const int kb = kk2 << 5;
        v16bf pf = frag_cat(&pS[wave][l16][kb + half * 8],
                            &pS[wave][l16][kb + 16 + half * 8]);
        #pragma unroll
        for (int dt = 0; dt < 2; ++dt) {
            const int d = (dt << 4) + l16;
            v16bf vf = frag_cat(&vS[d][kb + half * 8], &vS[d][kb + 16 + half * 8]);
            oacc[dt] = __builtin_amdgcn_wmma_f32_16x16x32_bf16(
                false, pf, false, vf, (short)0, oacc[dt], false, false);
        }
    }

    // ---- normalize by row sum and store bf16 (B_, N, h, d) -> (B_*N, 512) ----
    #pragma unroll
    for (int dt = 0; dt < 2; ++dt) {
        #pragma unroll
        for (int j = 0; j < 8; ++j) {
            const int n = rbase + half * 8 + j;
            if (n < NTOK) {
                const int d = (dt << 4) + l16;
                attn_out[(size_t)(b * NTOK + n) * DIM + head * HDIM + d] =
                    (bf16_t)(oacc[dt][j] * rinv[j]);
            }
        }
    }
}

// ---------------------------------------------------------------------------
extern "C" void kernel_launch(void* const* d_in, const int* in_sizes, int n_in,
                              void* d_out, int out_size, void* d_ws, size_t ws_size,
                              hipStream_t stream)
{
    const float* x      = (const float*)d_in[0];
    const float* w_qkv  = (const float*)d_in[1];
    const float* b_qkv  = (const float*)d_in[2];
    const float* rpb    = (const float*)d_in[3];
    const float* w_proj = (const float*)d_in[4];
    const float* b_proj = (const float*)d_in[5];
    const int*   relidx = (const int*)d_in[6];

    bf16_t* qkv_ws = (bf16_t*)d_ws;                             // 100352 x 1536 bf16
    const size_t qkv_bytes = (size_t)MTOT * (3 * DIM) * sizeof(bf16_t);
    bf16_t* attn_ws = (bf16_t*)((char*)d_ws + qkv_bytes);       // 100352 x 512 bf16
    const size_t attn_bytes = (size_t)MTOT * DIM * sizeof(bf16_t);
    float* bias_ws = (float*)((char*)d_ws + qkv_bytes + attn_bytes); // 16x64x64 f32

    // 0) padded relative-position bias table (tiny, L2-resident)
    bias_pad_kernel<<<HEADS, 256, 0, stream>>>(rpb, relidx, bias_ws);

    // 1) QKV projection: (100352 x 512) * (512 x 1536) + b_qkv  -> bf16 ws
    gemm_bf16_wmma<false, true><<<dim3(12, 784), 256, 0, stream>>>(
        (const void*)x, w_qkv, b_qkv, (void*)qkv_ws, 3 * DIM, DIM);

    // 2) windowed multi-head attention with relative position bias
    attn_wmma<<<BWIN * HEADS, 128, 0, stream>>>(qkv_ws, bias_ws, attn_ws);

    // 3) output projection: (100352 x 512) * (512 x 512) + b_proj -> f32 out
    gemm_bf16_wmma<true, false><<<dim3(4, 784), 256, 0, stream>>>(
        (const void*)attn_ws, w_proj, b_proj, d_out, DIM, DIM);
}
